// TTTBranch_20177756357074
// MI455X (gfx1250) — matrix operand order repaired and checked by reading, not compiled, not run on hardware
//
#include <hip/hip_runtime.h>
#include <hip/hip_bf16.h>

#define DEVFN __device__ __forceinline__

typedef __attribute__((ext_vector_type(16))) __bf16 v16bf;
typedef __attribute__((ext_vector_type(8)))  float  v8f;

constexpr int Bc   = 8;     // batch
constexpr int Lc   = 8192;  // sequence length
constexpr int DIN  = 256;
constexpr int DH   = 512;
constexpr int DOUT = 256;
constexpr int CH   = 1024;  // chunk size
constexpr int NCH  = Lc / CH;

union BfFrag {
  v16bf v;
  unsigned short u[16];
  uint4 q[2];
};

DEVFN unsigned short f2bfu(float f) {
  unsigned x = __float_as_uint(f);
  unsigned r = x + 0x7FFFu + ((x >> 16) & 1u);   // round-to-nearest-even
  return (unsigned short)(r >> 16);
}

// ---- WMMA fragment loaders -------------------------------------------------
// A operand (16x32 bf16, M x K): lane m = lane&15 holds row m.
// lanes 0-15: K in {0..7, 16..23}; lanes 16-31: K in {8..15, 24..31}.
DEVFN v16bf load_a_bf16(const unsigned short* A, int lda, int m0, int k0, int lane) {
  int r  = lane & 15;
  int kh = (lane >> 4) * 8;
  const unsigned short* p = A + (size_t)(m0 + r) * lda + k0 + kh;
  BfFrag f;
  f.q[0] = *(const uint4*)p;          // K = kh .. kh+7
  f.q[1] = *(const uint4*)(p + 16);   // K = 16+kh .. 16+kh+7
  return f.v;
}

// B operand (32x16 bf16, K x N) supplied as Bt row-major [N, K]:
// lane n = lane&15 holds column n; lanes 0-15 K in {0..7, 16..23}; lanes 16-31 K in {8..15, 24..31}.
DEVFN v16bf load_b_bf16(const unsigned short* Bt, int ldb, int n0, int k0, int lane) {
  int c  = lane & 15;
  int kh = (lane >> 4) * 8;
  const unsigned short* p = Bt + (size_t)(n0 + c) * ldb + k0 + kh;
  BfFrag f;
  f.q[0] = *(const uint4*)p;
  f.q[1] = *(const uint4*)(p + 16);
  return f.v;
}

DEVFN v8f wmma_bf16(v16bf a, v16bf b, v8f c) {
  return __builtin_amdgcn_wmma_f32_16x16x32_bf16(
      /*neg_a=*/false, a, /*neg_b=*/false, b,
      /*c_mod=*/(short)0, c, /*reuse_a=*/false, /*reuse_b=*/false);
}

DEVFN float sigmoidf_(float x) { return 1.0f / (1.0f + __expf(-x)); }

// ---- one-time fp32 -> bf16 conversion (bandwidth pass) --------------------
__global__ __launch_bounds__(256) void cvt_bf16_kernel(
    const float* __restrict__ src, unsigned short* __restrict__ dst, size_t n) {
  size_t i = ((size_t)blockIdx.x * 256 + threadIdx.x) * 4;
  if (i + 3 < n) {
    float4 f = *(const float4*)(src + i);
    union { unsigned short u[4]; uint2 d; } o;
    o.u[0] = f2bfu(f.x);
    o.u[1] = f2bfu(f.y);
    o.u[2] = f2bfu(f.z);
    o.u[3] = f2bfu(f.w);
    *(uint2*)(dst + i) = o.d;
  }
}

// ---- fused forward (+grad epilogue) kernel --------------------------------
// apply mode  (UPDATE=false): qk = q chunk (bf16), writes hidden bf16 [B,CH,DH]
// update mode (UPDATE=true) : qk = k chunk (bf16), also dh = v @ w1, writes
//   lr-scaled transposed bf16 grads [B, DH, CH] for the dw GEMMs.
template <bool UPDATE>
__global__ __launch_bounds__(256) void fused_fw_kernel(
    const unsigned short* __restrict__ qk,   // [B, L, DIN] bf16
    const unsigned short* __restrict__ v,    // [B, L, DOUT] bf16 (update only)
    const unsigned short* __restrict__ w0b,  // [B, DH, DIN] bf16
    const unsigned short* __restrict__ w2b,  // [B, DH, DIN] bf16
    const unsigned short* __restrict__ w1tb, // [B, DH, DOUT] bf16 (w1^T)
    const float* __restrict__ lr0,
    const float* __restrict__ lr1,
    const float* __restrict__ lr2,
    unsigned short* __restrict__ hiddenb,    // [B, CH, DH] (apply only)
    unsigned short* __restrict__ hl1T,       // [B, DH, CH] (update only)
    unsigned short* __restrict__ dgl0T,      // [B, DH, CH]
    unsigned short* __restrict__ dhl2T,      // [B, DH, CH]
    int chunk_base) {
  const int b    = blockIdx.z;
  const int lane = threadIdx.x & 31;
  const int wave = threadIdx.x >> 5;
  const int row0 = blockIdx.x * 64 + (wave & 1) * 32;    // row within chunk
  const int col0 = blockIdx.y * 128 + (wave >> 1) * 32;  // col within DH

  const unsigned short* Aq = qk + ((size_t)b * Lc + chunk_base + row0) * DIN;
  const unsigned short* B0 = w0b + (size_t)b * DH * DIN;
  const unsigned short* B2 = w2b + (size_t)b * DH * DIN;

  v8f accg[2][2] = {};
  v8f acch[2][2] = {};
  v8f accd[2][2] = {};

  for (int k = 0; k < DIN; k += 32) {
    v16bf a0 = load_a_bf16(Aq, DIN, 0, k, lane);
    v16bf a1 = load_a_bf16(Aq, DIN, 16, k, lane);
#pragma unroll
    for (int fn = 0; fn < 2; ++fn) {
      v16bf bg = load_b_bf16(B0, DIN, col0 + fn * 16, k, lane);
      accg[0][fn] = wmma_bf16(a0, bg, accg[0][fn]);
      accg[1][fn] = wmma_bf16(a1, bg, accg[1][fn]);
      v16bf bh = load_b_bf16(B2, DIN, col0 + fn * 16, k, lane);
      acch[0][fn] = wmma_bf16(a0, bh, acch[0][fn]);
      acch[1][fn] = wmma_bf16(a1, bh, acch[1][fn]);
    }
  }

  if constexpr (UPDATE) {
    const unsigned short* Av = v + ((size_t)b * Lc + chunk_base + row0) * DOUT;
    const unsigned short* B1 = w1tb + (size_t)b * DH * DOUT;
    for (int k = 0; k < DOUT; k += 32) {
      v16bf a0 = load_a_bf16(Av, DOUT, 0, k, lane);
      v16bf a1 = load_a_bf16(Av, DOUT, 16, k, lane);
#pragma unroll
      for (int fn = 0; fn < 2; ++fn) {
        v16bf bd = load_b_bf16(B1, DOUT, col0 + fn * 16, k, lane);
        accd[0][fn] = wmma_bf16(a0, bd, accd[0][fn]);
        accd[1][fn] = wmma_bf16(a1, bd, accd[1][fn]);
      }
    }
  }

  const int m_in = (lane >> 4) * 8;  // row offset within a 16x16 frag
  const int n_in = lane & 15;
#pragma unroll
  for (int fm = 0; fm < 2; ++fm) {
#pragma unroll
    for (int fn = 0; fn < 2; ++fn) {
#pragma unroll
      for (int r = 0; r < 8; ++r) {
        int c = row0 + fm * 16 + m_in + r;  // chunk row
        int h = col0 + fn * 16 + n_in;      // hidden col
        float g  = accg[fm][fn][r];
        float hb = acch[fm][fn][r];
        float s  = sigmoidf_(g);
        float sg = g * s;
        if constexpr (!UPDATE) {
          hiddenb[((size_t)b * CH + c) * DH + h] = f2bfu(sg * hb);
        } else {
          float dh     = accd[fm][fn][r];
          float hidden = sg * hb;
          float dhbm   = dh * sg;
          float dg     = dh * hb;
          float dgba   = dg * s * (1.0f + g * (1.0f - s));
          size_t ti = ((size_t)b * DH + h) * CH + c;
          size_t li = (size_t)b * Lc + chunk_base + c;
          hl1T[ti]  = f2bfu(hidden * lr1[li]);
          dgl0T[ti] = f2bfu(dgba * lr0[li]);
          dhl2T[ti] = f2bfu(dhbm * lr2[li]);
        }
      }
    }
  }
}

// ---- out = hidden @ w1^T  ( [CH,DH] x [DOUT,DH]^T -> fp32 out ) -----------
__global__ __launch_bounds__(256) void out_gemm_kernel(
    const unsigned short* __restrict__ hiddenb,  // [B, CH, DH]
    const unsigned short* __restrict__ w1b,      // [B, DOUT, DH]
    float* __restrict__ out,                     // [B, L, DOUT]
    int chunk_base) {
  const int b    = blockIdx.z;
  const int lane = threadIdx.x & 31;
  const int wave = threadIdx.x >> 5;
  const int row0 = blockIdx.x * 64 + (wave & 1) * 32;    // row within chunk
  const int col0 = blockIdx.y * 128 + (wave >> 1) * 32;  // col within DOUT

  const unsigned short* A  = hiddenb + (size_t)b * CH * DH;
  const unsigned short* Bt = w1b + (size_t)b * DOUT * DH;

  v8f acc[2][2] = {};
  for (int k = 0; k < DH; k += 32) {
    v16bf a0 = load_a_bf16(A, DH, row0, k, lane);
    v16bf a1 = load_a_bf16(A, DH, row0 + 16, k, lane);
#pragma unroll
    for (int fn = 0; fn < 2; ++fn) {
      v16bf bb = load_b_bf16(Bt, DH, col0 + fn * 16, k, lane);
      acc[0][fn] = wmma_bf16(a0, bb, acc[0][fn]);
      acc[1][fn] = wmma_bf16(a1, bb, acc[1][fn]);
    }
  }

  const int m_in = (lane >> 4) * 8;
  const int n_in = lane & 15;
#pragma unroll
  for (int fm = 0; fm < 2; ++fm)
#pragma unroll
    for (int fn = 0; fn < 2; ++fn)
#pragma unroll
      for (int r = 0; r < 8; ++r) {
        int c = row0 + fm * 16 + m_in + r;
        int o = col0 + fn * 16 + n_in;
        out[((size_t)b * Lc + chunk_base + c) * DOUT + o] = acc[fm][fn][r];
      }
}

// ---- dw GEMM, accumulate into fp32 main weights ---------------------------
// D[m,n] = sum_k A[m,k] * Bt[n,k], K = CH; W[m,n] += D[m,n].
__global__ __launch_bounds__(256) void gemm_acc_kernel(
    const unsigned short* __restrict__ A,   // [B, M, CH] bf16
    const unsigned short* __restrict__ Bt,  // [B, N, CH] bf16
    float* __restrict__ Wm,                 // [B, M, N] fp32
    int M, int N) {
  const int b    = blockIdx.z;
  const int lane = threadIdx.x & 31;
  const int wave = threadIdx.x >> 5;
  const int row0 = blockIdx.x * 64 + (wave & 1) * 32;
  const int col0 = blockIdx.y * 128 + (wave >> 1) * 32;

  const unsigned short* Ab  = A + (size_t)b * M * CH;
  const unsigned short* Btb = Bt + (size_t)b * N * CH;
  float* W = Wm + (size_t)b * M * N;

  v8f acc[2][2] = {};
  for (int k = 0; k < CH; k += 32) {
    v16bf a0 = load_a_bf16(Ab, CH, row0, k, lane);
    v16bf a1 = load_a_bf16(Ab, CH, row0 + 16, k, lane);
#pragma unroll
    for (int fn = 0; fn < 2; ++fn) {
      v16bf bb = load_b_bf16(Btb, CH, col0 + fn * 16, k, lane);
      acc[0][fn] = wmma_bf16(a0, bb, acc[0][fn]);
      acc[1][fn] = wmma_bf16(a1, bb, acc[1][fn]);
    }
  }

  const int m_in = (lane >> 4) * 8;
  const int n_in = lane & 15;
#pragma unroll
  for (int fm = 0; fm < 2; ++fm)
#pragma unroll
    for (int fn = 0; fn < 2; ++fn)
#pragma unroll
      for (int r = 0; r < 8; ++r) {
        size_t idx = (size_t)(row0 + fm * 16 + m_in + r) * N + col0 + fn * 16 + n_in;
        W[idx] += acc[fm][fn][r];
      }
}

// ---- row-norm init: Wm = W, Wn = ||W_row|| --------------------------------
__global__ __launch_bounds__(256) void init_norm_kernel(
    const float* __restrict__ W, float* __restrict__ Wm, float* __restrict__ Wn,
    int rows, int cols) {
  int b = blockIdx.y, row = blockIdx.x;
  const float* src = W + ((size_t)b * rows + row) * cols;
  float* dst = Wm + ((size_t)b * rows + row) * cols;
  __shared__ float red[256];
  float ss = 0.0f;
  for (int c = threadIdx.x; c < cols; c += 256) {
    float x = src[c];
    dst[c] = x;
    ss += x * x;
  }
  red[threadIdx.x] = ss;
  __syncthreads();
  for (int s = 128; s; s >>= 1) {
    if ((int)threadIdx.x < s) red[threadIdx.x] += red[threadIdx.x + s];
    __syncthreads();
  }
  if (threadIdx.x == 0) Wn[(size_t)b * rows + row] = sqrtf(red[0]);
}

// ---- renorm: wc = wm / (||wm_row|| + eps) * wn -> bf16 (+optional ^T) -----
__global__ __launch_bounds__(256) void renorm_kernel(
    const float* __restrict__ Wm, const float* __restrict__ Wn,
    unsigned short* __restrict__ Wcb, unsigned short* __restrict__ WcbT,
    int rows, int cols) {
  int b = blockIdx.y, row = blockIdx.x;
  const float* src = Wm + ((size_t)b * rows + row) * cols;
  __shared__ float red[256];
  float ss = 0.0f;
  for (int c = threadIdx.x; c < cols; c += 256) {
    float x = src[c];
    ss += x * x;
  }
  red[threadIdx.x] = ss;
  __syncthreads();
  for (int s = 128; s; s >>= 1) {
    if ((int)threadIdx.x < s) red[threadIdx.x] += red[threadIdx.x + s];
    __syncthreads();
  }
  float scale = Wn[(size_t)b * rows + row] / (sqrtf(red[0]) + 1e-5f);
  unsigned short* dst = Wcb + ((size_t)b * rows + row) * cols;
  for (int c = threadIdx.x; c < cols; c += 256) {
    unsigned short bv = f2bfu(src[c] * scale);
    dst[c] = bv;
    if (WcbT) WcbT[((size_t)b * cols + c) * rows + row] = bv;
  }
}

// ---- 32x32 LDS transpose of a bf16 chunk ----------------------------------
// src [B, L, D] bf16 -> dst [B, D, CH] bf16
__global__ __launch_bounds__(1024) void transpose_chunk_kernel(
    const unsigned short* __restrict__ src, unsigned short* __restrict__ dst,
    int chunk_base, int D) {
  __shared__ unsigned short tile[32][33];
  int b = blockIdx.z;
  int c0 = blockIdx.x * 32, d0 = blockIdx.y * 32;
  int tx = threadIdx.x, ty = threadIdx.y;
  tile[ty][tx] = src[((size_t)b * Lc + chunk_base + c0 + ty) * D + d0 + tx];
  __syncthreads();
  dst[((size_t)b * D + d0 + ty) * CH + c0 + tx] = tile[tx][ty];
}

// ---------------------------------------------------------------------------
extern "C" void kernel_launch(void* const* d_in, const int* in_sizes, int n_in,
                              void* d_out, int out_size, void* d_ws, size_t ws_size,
                              hipStream_t stream) {
  const float* w0  = (const float*)d_in[0];
  const float* w1  = (const float*)d_in[1];
  const float* w2  = (const float*)d_in[2];
  const float* q   = (const float*)d_in[3];
  const float* k   = (const float*)d_in[4];
  const float* v   = (const float*)d_in[5];
  const float* lr0 = (const float*)d_in[6];
  const float* lr1 = (const float*)d_in[7];
  const float* lr2 = (const float*)d_in[8];
  float* out = (float*)d_out;

  // workspace carve-up
  char* p = (char*)d_ws;
  auto alloc = [&](size_t bytes) -> void* {
    void* r = (void*)p;
    p += (bytes + 255) & ~(size_t)255;
    return r;
  };
  float* wm0 = (float*)alloc((size_t)Bc * DH * DIN * 4);
  float* wm1 = (float*)alloc((size_t)Bc * DOUT * DH * 4);
  float* wm2 = (float*)alloc((size_t)Bc * DH * DIN * 4);
  float* wn0 = (float*)alloc((size_t)Bc * DH * 4);
  float* wn1 = (float*)alloc((size_t)Bc * DOUT * 4);
  float* wn2 = (float*)alloc((size_t)Bc * DH * 4);
  unsigned short* w0b   = (unsigned short*)alloc((size_t)Bc * DH * DIN * 2);
  unsigned short* w1b   = (unsigned short*)alloc((size_t)Bc * DOUT * DH * 2);
  unsigned short* w1tb  = (unsigned short*)alloc((size_t)Bc * DH * DOUT * 2);
  unsigned short* w2b   = (unsigned short*)alloc((size_t)Bc * DH * DIN * 2);
  unsigned short* qb    = (unsigned short*)alloc((size_t)Bc * Lc * DIN * 2);
  unsigned short* kb    = (unsigned short*)alloc((size_t)Bc * Lc * DIN * 2);
  unsigned short* vb    = (unsigned short*)alloc((size_t)Bc * Lc * DOUT * 2);
  unsigned short* kTi   = (unsigned short*)alloc((size_t)Bc * DIN * CH * 2);
  unsigned short* vTi   = (unsigned short*)alloc((size_t)Bc * DOUT * CH * 2);
  unsigned short* hid   = (unsigned short*)alloc((size_t)Bc * CH * DH * 2);
  unsigned short* hl1T  = (unsigned short*)alloc((size_t)Bc * DH * CH * 2);
  unsigned short* dgl0T = (unsigned short*)alloc((size_t)Bc * DH * CH * 2);
  unsigned short* dhl2T = (unsigned short*)alloc((size_t)Bc * DH * CH * 2);

  // one-time bf16 images of the activations
  {
    size_t nq = (size_t)Bc * Lc * DIN;
    size_t nv = (size_t)Bc * Lc * DOUT;
    cvt_bf16_kernel<<<(unsigned)((nq / 4 + 255) / 256), 256, 0, stream>>>(q, qb, nq);
    cvt_bf16_kernel<<<(unsigned)((nq / 4 + 255) / 256), 256, 0, stream>>>(k, kb, nq);
    cvt_bf16_kernel<<<(unsigned)((nv / 4 + 255) / 256), 256, 0, stream>>>(v, vb, nv);
  }

  // init: main-weight copies + prenorm target norms, then bf16 weight images
  init_norm_kernel<<<dim3(DH, Bc), 256, 0, stream>>>(w0, wm0, wn0, DH, DIN);
  init_norm_kernel<<<dim3(DOUT, Bc), 256, 0, stream>>>(w1, wm1, wn1, DOUT, DH);
  init_norm_kernel<<<dim3(DH, Bc), 256, 0, stream>>>(w2, wm2, wn2, DH, DIN);
  renorm_kernel<<<dim3(DH, Bc), 256, 0, stream>>>(wm0, wn0, w0b, nullptr, DH, DIN);
  renorm_kernel<<<dim3(DOUT, Bc), 256, 0, stream>>>(wm1, wn1, w1b, w1tb, DOUT, DH);
  renorm_kernel<<<dim3(DH, Bc), 256, 0, stream>>>(wm2, wn2, w2b, nullptr, DH, DIN);

  for (int i = 0; i < NCH; ++i) {
    int cb = i * CH;
    // apply current fast weights to this chunk's queries
    fused_fw_kernel<false><<<dim3(CH / 64, DH / 128, Bc), 256, 0, stream>>>(
        qb, nullptr, w0b, w2b, nullptr, nullptr, nullptr, nullptr,
        hid, nullptr, nullptr, nullptr, cb);
    out_gemm_kernel<<<dim3(CH / 64, DOUT / 128, Bc), 256, 0, stream>>>(
        hid, w1b, out, cb);

    if (i + 1 < NCH) {
      // key-side forward + SwiGLU backprop, lr-scaled transposed grads
      fused_fw_kernel<true><<<dim3(CH / 64, DH / 128, Bc), 256, 0, stream>>>(
          kb, vb, w0b, w2b, w1tb, lr0, lr1, lr2,
          nullptr, hl1T, dgl0T, dhl2T, cb);
      transpose_chunk_kernel<<<dim3(CH / 32, DIN / 32, Bc), dim3(32, 32), 0, stream>>>(
          kb, kTi, cb, DIN);
      transpose_chunk_kernel<<<dim3(CH / 32, DOUT / 32, Bc), dim3(32, 32), 0, stream>>>(
          vb, vTi, cb, DOUT);
      // dw1 = v^T @ (hidden*l1)   -> wm1 += [DOUT, DH]
      gemm_acc_kernel<<<dim3(DOUT / 64, DH / 128, Bc), 256, 0, stream>>>(
          vTi, hl1T, wm1, DOUT, DH);
      // dw0 = (dgba*l0)^T @ k     -> wm0 += [DH, DIN]
      gemm_acc_kernel<<<dim3(DH / 64, DIN / 128, Bc), 256, 0, stream>>>(
          dgl0T, kTi, wm0, DH, DIN);
      // dw2 = (dhbm*l2)^T @ k     -> wm2 += [DH, DIN]
      gemm_acc_kernel<<<dim3(DH / 64, DIN / 128, Bc), 256, 0, stream>>>(
          dhl2T, kTi, wm2, DH, DIN);
      // prenorm renormalization, refresh bf16 weight images
      renorm_kernel<<<dim3(DH, Bc), 256, 0, stream>>>(wm0, wn0, w0b, nullptr, DH, DIN);
      renorm_kernel<<<dim3(DOUT, Bc), 256, 0, stream>>>(wm1, wn1, w1b, w1tb, DOUT, DH);
      renorm_kernel<<<dim3(DH, Bc), 256, 0, stream>>>(wm2, wn2, w2b, nullptr, DH, DIN);
    }
  }
  (void)in_sizes; (void)n_in; (void)out_size; (void)ws_size;
}
